// ARModel_37752762532065
// MI455X (gfx1250) — matrix operand order repaired and checked by reading, not compiled, not run on hardware
//
#include <hip/hip_runtime.h>
#include <math.h>

// ---------------- CDNA5 WMMA types ----------------
typedef _Float16 h8  __attribute__((ext_vector_type(8)));
typedef _Float16 h16 __attribute__((ext_vector_type(16)));
typedef float    f8  __attribute__((ext_vector_type(8)));
union H16 { h16 v; h8 p[2]; };

// ---------------- model dims ----------------
#define B_    8
#define T_    640
#define MEL_  229
#define CU_   48
#define C3_   96
#define MW2_  114
#define MW3_  57
#define FCN_  768
#define H_    768
#define G4_   3072
#define NC_   5
#define KCONV_ 448    // 9*48 = 432 padded to 448 (mult of 32)
#define KFC_   5472   // 96*57 (already mult of 32)
#define KIN0_  2496   // 2480 padded
#define NOUT_  440
#define BT_   (B_*T_)

// ---------------- CDNA5 async global->LDS helpers --------------------------
__device__ __forceinline__ void async_b128(unsigned int lds, const _Float16* g) {
  asm volatile("global_load_async_to_lds_b128 %0, %1, off"
               :: "v"(lds), "v"((unsigned long long)(uintptr_t)g) : "memory");
}
__device__ __forceinline__ void wait_asynccnt0() {
  asm volatile("s_wait_asynccnt 0x0" ::: "memory");
}
__device__ __forceinline__ void wait_asynccnt3() {
  asm volatile("s_wait_asynccnt 0x3" ::: "memory");
}

// ---------------- WMMA GEMM: C[M,N] = A[M,K] * W[N,K]^T (+bias, relu) -------
// Block tile 128(M) x 64(N), BK=32. 8 wave32 waves arranged 4(M)x2(N); each
// wave computes a 32x32 quad (2x2 tiles of 16x16) -> 4 v_wmma per K-step.
#define BM 128
#define BN 64
#define BK 32
#define LDT (BK + 8)   // pad LDS rows: 40 halves = 80B -> avoids bank reuse

// Guarded kernel, templated on CONV (implicit im2col gather vs plain rows).
template <bool CONV>
__global__ __launch_bounds__(256) void wmma_gemm_kernel(
    const _Float16* __restrict__ A, const _Float16* __restrict__ W,
    const float* __restrict__ bias, float* __restrict__ outF,
    _Float16* __restrict__ outH, int Mrows, int N, int K, int relu,
    const _Float16* __restrict__ convIn, int convT, int convMw, int Cin)
{
  __shared__ __align__(16) _Float16 As[BM][LDT];
  __shared__ __align__(16) _Float16 Ws[BN][LDT];

  const int tid  = threadIdx.x;
  const int lane = tid & 31;
  const int wave = tid >> 5;
  const int hsel = lane >> 4;
  const int l15  = lane & 15;
  const int wm   = wave >> 1;
  const int wn   = wave & 1;

  const int row0 = blockIdx.y * BM;
  const int n0   = blockIdx.x * BN;

  f8 zf = {};
  f8 acc[2][2];
  acc[0][0] = zf; acc[0][1] = zf; acc[1][0] = zf; acc[1][1] = zf;

  const int kReal = CONV ? 9 * Cin : K;
  const int TMw   = CONV ? convT * convMw : 0;

  for (int k0 = 0; k0 < K; k0 += BK) {
    // ---- W tile: 64 rows x 32 halves (one 16B chunk per thread)
    {
      int r = tid >> 2, cc = tid & 3;
      int n = n0 + r;
      h8 val = {};
      if (n < N) val = *(const h8*)(W + (size_t)n * K + k0 + cc * 8);
      *(h8*)&Ws[r][cc * 8] = val;
    }
    // ---- A tile: 128 rows x 32 halves (two 16B chunks per thread)
    #pragma unroll
    for (int c = 0; c < 2; ++c) {
      int q = tid + c * 256;
      int r = q >> 2, cc = q & 3;
      int gRow = row0 + r;
      h8 val = {};
      if (gRow < Mrows) {
        if constexpr (!CONV) {
          val = *(const h8*)(A + (size_t)gRow * K + k0 + cc * 8);
        } else {
          // implicit im2col: row = pixel (b,t,m); k = (kh*3+kw)*Cin + ci
          int b = gRow / TMw; int rem = gRow - b * TMw;
          int t = rem / convMw; int m = rem - t * convMw;
          int kk = k0 + cc * 8;
          if (kk < kReal) {
            int kh = kk / (3 * Cin); int r2 = kk - kh * 3 * Cin;
            int kw = r2 / Cin;       int ci = r2 - kw * Cin;
            #pragma unroll
            for (int j = 0; j < 8; ++j) {
              _Float16 x = (_Float16)0;
              if (kk + j < kReal) {
                int tt = t + kh - 1, mm = m + kw - 1;
                if (tt >= 0 && tt < convT && mm >= 0 && mm < convMw)
                  x = convIn[((size_t)(b * convT + tt) * convMw + mm) * Cin + ci];
              }
              val[j] = x;
              if (++ci == Cin) { ci = 0; if (++kw == 3) { kw = 0; ++kh; } }
            }
          }
        }
      }
      *(h8*)&As[r][cc * 8] = val;
    }
    __syncthreads();

    H16 af[2], bf[2];
    #pragma unroll
    for (int mi = 0; mi < 2; ++mi) {
      const _Float16* src = &As[wm * 32 + mi * 16 + l15][hsel * 8];
      af[mi].p[0] = *(const h8*)src;
      af[mi].p[1] = *(const h8*)(src + 16);
    }
    #pragma unroll
    for (int ni = 0; ni < 2; ++ni) {
      const _Float16* src = &Ws[wn * 32 + ni * 16 + l15][hsel * 8];
      bf[ni].p[0] = *(const h8*)src;
      bf[ni].p[1] = *(const h8*)(src + 16);
    }
    #pragma unroll
    for (int mi = 0; mi < 2; ++mi)
      #pragma unroll
      for (int ni = 0; ni < 2; ++ni)
        acc[mi][ni] = __builtin_amdgcn_wmma_f32_16x16x32_f16(
            false, af[mi].v, false, bf[ni].v, (short)0, acc[mi][ni], false, false);
    __syncthreads();
  }

  #pragma unroll
  for (int mi = 0; mi < 2; ++mi)
    #pragma unroll
    for (int ni = 0; ni < 2; ++ni)
      #pragma unroll
      for (int r = 0; r < 8; ++r) {
        int gM = row0 + wm * 32 + mi * 16 + r + hsel * 8;
        int gN = n0 + wn * 32 + ni * 16 + l15;
        if (gM < Mrows && gN < N) {
          float v = acc[mi][ni][r];
          if (bias) v += bias[gN];
          if (relu) v = fmaxf(v, 0.f);
          if (outF) outF[(size_t)gM * N + gN] = v;
          if (outH) outH[(size_t)gM * N + gN] = (_Float16)v;
        }
      }
}

// Aligned fast path (M%128==0, N%64==0, K%32==0): double-buffered LDS tiles
// filled by global_load_async_to_lds_b128, pipelined one K-step ahead and
// synchronized with s_wait_asynccnt (async ops retire in order, so waiting
// for <=3 outstanding guarantees the current tile's 3 transfers landed).
__global__ __launch_bounds__(256) void wmma_gemm_async_kernel(
    const _Float16* __restrict__ A, const _Float16* __restrict__ W,
    const float* __restrict__ bias, float* __restrict__ outF,
    _Float16* __restrict__ outH, int Mrows, int N, int K, int relu)
{
  __shared__ __align__(16) _Float16 As[2][BM][LDT];
  __shared__ __align__(16) _Float16 Ws[2][BN][LDT];

  const int tid  = threadIdx.x;
  const int lane = tid & 31;
  const int wave = tid >> 5;
  const int hsel = lane >> 4;
  const int l15  = lane & 15;
  const int wm   = wave >> 1;
  const int wn   = wave & 1;
  const int row0 = blockIdx.y * BM;
  const int n0   = blockIdx.x * BN;

  const int rq = tid >> 2, cc = tid & 3;
  const _Float16* gW  = W + (size_t)(n0 + rq) * K + cc * 8;
  const _Float16* gA0 = A + (size_t)(row0 + rq) * K + cc * 8;
  const _Float16* gA1 = A + (size_t)(row0 + 64 + rq) * K + cc * 8;
  const unsigned int lW  = (unsigned int)(uintptr_t)&Ws[0][rq][cc * 8];
  const unsigned int lA0 = (unsigned int)(uintptr_t)&As[0][rq][cc * 8];
  const unsigned int lA1 = (unsigned int)(uintptr_t)&As[0][64 + rq][cc * 8];
  const unsigned int wsStep = (unsigned int)(sizeof(_Float16) * BN * LDT);
  const unsigned int asStep = (unsigned int)(sizeof(_Float16) * BM * LDT);

  f8 zf = {};
  f8 acc[2][2];
  acc[0][0] = zf; acc[0][1] = zf; acc[1][0] = zf; acc[1][1] = zf;

  const int nk = K / BK;
  // prologue: tile 0 -> buffer 0
  async_b128(lW,  gW);
  async_b128(lA0, gA0);
  async_b128(lA1, gA1);

  for (int ki = 0; ki < nk; ++ki) {
    const int cur = ki & 1;
    if (ki + 1 < nk) {
      const unsigned int nxt = cur ^ 1;
      const size_t ko = (size_t)(ki + 1) * BK;
      async_b128(lW  + nxt * wsStep, gW  + ko);
      async_b128(lA0 + nxt * asStep, gA0 + ko);
      async_b128(lA1 + nxt * asStep, gA1 + ko);
      wait_asynccnt3();     // current tile's 3 transfers have retired
    } else {
      wait_asynccnt0();
    }
    __syncthreads();        // all waves' tile-ki transfers visible in LDS

    H16 af[2], bf[2];
    #pragma unroll
    for (int mi = 0; mi < 2; ++mi) {
      const _Float16* src = &As[cur][wm * 32 + mi * 16 + l15][hsel * 8];
      af[mi].p[0] = *(const h8*)src;          // k: 0-7  / 8-15
      af[mi].p[1] = *(const h8*)(src + 16);   // k: 16-23/ 24-31
    }
    #pragma unroll
    for (int ni = 0; ni < 2; ++ni) {
      const _Float16* src = &Ws[cur][wn * 32 + ni * 16 + l15][hsel * 8];
      bf[ni].p[0] = *(const h8*)src;          // k: 0-7  / 8-15
      bf[ni].p[1] = *(const h8*)(src + 16);   // k: 16-23/ 24-31
    }
    #pragma unroll
    for (int mi = 0; mi < 2; ++mi)
      #pragma unroll
      for (int ni = 0; ni < 2; ++ni)
        acc[mi][ni] = __builtin_amdgcn_wmma_f32_16x16x32_f16(
            false, af[mi].v, false, bf[ni].v, (short)0, acc[mi][ni], false, false);
    __syncthreads();        // reads of buf[cur] done before it is refilled
  }

  #pragma unroll
  for (int mi = 0; mi < 2; ++mi)
    #pragma unroll
    for (int ni = 0; ni < 2; ++ni)
      #pragma unroll
      for (int r = 0; r < 8; ++r) {
        int gM = row0 + wm * 32 + mi * 16 + r + hsel * 8;
        int gN = n0 + wn * 32 + ni * 16 + l15;
        float v = acc[mi][ni][r];
        if (bias) v += bias[gN];
        if (relu) v = fmaxf(v, 0.f);
        if (outF) outF[(size_t)gM * N + gN] = v;
        if (outH) outH[(size_t)gM * N + gN] = (_Float16)v;
      }
}

// ---------------- fused LSTM recurrent step --------------------------------
// g = gx[:,t,:] + h @ w_hh^T ; gates + cell update fused. B=8 padded to 16
// rows (rows 8..15 stay zero). 6 blocks x 8 waves; each wave owns 16 hidden
// units and computes i,f,g,o tiles (4 WMMA accumulators, K=768 -> 24 steps).
// w_hh (4.7MB fp16) stays resident in the 192MB L2 across the 640 steps.
#define HLD (H_ + 8)
__device__ __forceinline__ float sigmoidf_(float x) { return 1.f / (1.f + expf(-x)); }

__global__ __launch_bounds__(256) void lstm_step_kernel(
    const _Float16* __restrict__ hin, const _Float16* __restrict__ whh,
    const float* __restrict__ gx, float* __restrict__ c,
    _Float16* __restrict__ hout, _Float16* __restrict__ hs, int t)
{
  __shared__ __align__(16) _Float16 hls[16][HLD];
  const int tid  = threadIdx.x;
  const int lane = tid & 31;
  const int wave = tid >> 5;
  const int hsel = lane >> 4;
  const int l15  = lane & 15;

  for (int q = tid; q < 16 * (H_ / 8); q += 256) {
    int r = q / (H_ / 8), cc = q - r * (H_ / 8);
    *(h8*)&hls[r][cc * 8] = *(const h8*)(hin + (size_t)r * H_ + cc * 8);
  }
  __syncthreads();

  const int j = (blockIdx.x * 8 + wave) * 16;
  f8 zf = {};
  f8 acc[4]; acc[0] = zf; acc[1] = zf; acc[2] = zf; acc[3] = zf;

  for (int kk = 0; kk < H_; kk += 32) {
    H16 a;
    const _Float16* asrc = &hls[l15][kk + hsel * 8];
    a.p[0] = *(const h8*)asrc;
    a.p[1] = *(const h8*)(asrc + 16);
    #pragma unroll
    for (int g = 0; g < 4; ++g) {
      H16 b;
      const _Float16* bsrc = whh + (size_t)(g * H_ + j + l15) * H_ + kk + hsel * 8;
      b.p[0] = *(const h8*)bsrc;
      b.p[1] = *(const h8*)(bsrc + 16);
      acc[g] = __builtin_amdgcn_wmma_f32_16x16x32_f16(
          false, a.v, false, b.v, (short)0, acc[g], false, false);
    }
  }

  if (hsel == 0) {
    #pragma unroll
    for (int r = 0; r < 8; ++r) {
      int m = r;
      int n = j + l15;
      const float* gxr = gx + ((size_t)m * T_ + t) * G4_;
      float iv = acc[0][r] + gxr[n];
      float fv = acc[1][r] + gxr[H_ + n];
      float gv = acc[2][r] + gxr[2 * H_ + n];
      float ov = acc[3][r] + gxr[3 * H_ + n];
      float ig = sigmoidf_(iv), fg = sigmoidf_(fv);
      float gg = tanhf(gv),     og = sigmoidf_(ov);
      float cn = fg * c[m * H_ + n] + ig * gg;
      c[m * H_ + n] = cn;
      float h = og * tanhf(cn);
      hout[(size_t)m * H_ + n] = (_Float16)h;
      hs[((size_t)m * T_ + t) * H_ + n] = (_Float16)h;
    }
  }
}

// ---------------- small elementwise / prep kernels -------------------------
__global__ void logmel_kernel(const float* __restrict__ mel, _Float16* __restrict__ x) {
  int i = blockIdx.x * 256 + threadIdx.x;
  if (i >= BT_ * MEL_) return;
  x[i] = (_Float16)logf(fmaxf(mel[i], 1e-9f));
}

__global__ void conv_prep_kernel(const float* __restrict__ w, const float* __restrict__ cb,
    const float* __restrict__ g, const float* __restrict__ bb,
    const float* __restrict__ bm, const float* __restrict__ bv,
    _Float16* __restrict__ we, float* __restrict__ be, int O, int I, int kdst)
{
  int i = blockIdx.x * 256 + threadIdx.x;
  if (i >= O * kdst) return;
  int o = i / kdst, k = i - o * kdst;
  float s = g[o] * rsqrtf(bv[o] + 1e-5f);
  if (k == 0) be[o] = (cb[o] - bm[o]) * s + bb[o];
  float val = 0.f;
  if (k < 9 * I) {
    int kh = k / (3 * I); int r = k - kh * 3 * I;
    int kw = r / I;       int ci = r - kw * I;
    val = w[((o * I + ci) * 3 + kh) * 3 + kw] * s;  // OIHW -> [o][kh][kw][ci]
  }
  we[i] = (_Float16)val;
}

__global__ void cvtpad_kernel(const float* __restrict__ src, _Float16* __restrict__ dst,
                              int rows, int ksrc, int kdst) {
  int i = blockIdx.x * 256 + threadIdx.x;
  if (i >= rows * kdst) return;
  int r = i / kdst, k = i - r * kdst;
  dst[i] = (_Float16)((k < ksrc) ? src[(size_t)r * ksrc + k] : 0.f);
}

__global__ void biassum_kernel(const float* a, const float* b, float* o, int n) {
  int i = blockIdx.x * 256 + threadIdx.x;
  if (i < n) o[i] = a[i] + b[i];
}

__global__ void zero_kernel(unsigned int* p, int n) {
  int i = blockIdx.x * 256 + threadIdx.x;
  if (i < n) p[i] = 0u;
}

__global__ void conv1_kernel(const _Float16* __restrict__ x, const _Float16* __restrict__ w,
                             const float* __restrict__ be, _Float16* __restrict__ out) {
  int i = blockIdx.x * 256 + threadIdx.x;
  if (i >= BT_ * MEL_ * CU_) return;
  int c = i % CU_; int pm = i / CU_;
  int m = pm % MEL_; int bt = pm / MEL_;
  int t = bt % T_;  int b0 = bt - t;       // b*T
  float acc = be[c];
  #pragma unroll
  for (int kh = 0; kh < 3; kh++) {
    int tt = t + kh - 1;
    if (tt < 0 || tt >= T_) continue;
    #pragma unroll
    for (int kw = 0; kw < 3; kw++) {
      int mm = m + kw - 1;
      if (mm < 0 || mm >= MEL_) continue;
      acc += (float)x[(size_t)(b0 + tt) * MEL_ + mm] * (float)w[c * 9 + kh * 3 + kw];
    }
  }
  out[i] = (_Float16)fmaxf(acc, 0.f);
}

__global__ void pool2_kernel(const _Float16* __restrict__ in, _Float16* __restrict__ out) {
  int i = blockIdx.x * 256 + threadIdx.x;
  if (i >= BT_ * MW2_ * CU_) return;
  int c = i % CU_; int pm = i / CU_;
  int m = pm % MW2_; int bt = pm / MW2_;
  float a = (float)in[((size_t)bt * MEL_ + 2 * m) * CU_ + c];
  float b = (float)in[((size_t)bt * MEL_ + 2 * m + 1) * CU_ + c];
  out[i] = (_Float16)fmaxf(a, b);
}

// pool conv3 output AND reorder to FC layout: feature f = c*57 + m
__global__ void pool3_kernel(const _Float16* __restrict__ in, _Float16* __restrict__ out) {
  int i = blockIdx.x * 256 + threadIdx.x;
  if (i >= BT_ * KFC_) return;
  int f = i % KFC_; int bt = i / KFC_;
  int c = f / MW3_; int m = f - c * MW3_;
  float a = (float)in[((size_t)bt * MW2_ + 2 * m) * C3_ + c];
  float b = (float)in[((size_t)bt * MW2_ + 2 * m + 1) * C3_ + c];
  out[i] = (_Float16)fmaxf(a, b);
}

// comb[b,t,:] = [ emb(prev_label) (176) | ctx-stacked ac (2304, f=c*3+k) | pad ]
__global__ void comb_kernel(const int* __restrict__ plabel, const float* __restrict__ emb,
                            const _Float16* __restrict__ ac, _Float16* __restrict__ comb) {
  int i = blockIdx.x * 256 + threadIdx.x;
  if (i >= BT_ * KIN0_) return;
  int f = i % KIN0_; int bt = i / KIN0_;
  int t = bt % T_;  int b = bt / T_;
  float v = 0.f;
  if (f < 176) {
    int k = f >> 1, comp = f & 1;
    int lbl = plabel[(size_t)bt * 88 + k];
    v = emb[lbl * 2 + comp];
  } else if (f < 176 + FCN_ * 3) {
    int ff = f - 176;
    int c = ff / 3, k = ff - 3 * c;
    int tt = t + k - 1;
    if (tt >= 0 && tt < T_) v = (float)ac[((size_t)b * T_ + tt) * FCN_ + c];
  }
  comb[i] = (_Float16)v;
}

__global__ void logsoftmax_kernel(const float* __restrict__ logits, float* __restrict__ out) {
  int i = blockIdx.x * 256 + threadIdx.x;
  if (i >= BT_ * 88) return;
  int p = i % 88; int bt = i / 88;
  int t = bt % T_; int b = bt / T_;
  const float* src = logits + (size_t)bt * NOUT_ + p * NC_;
  float mx = src[0];
  #pragma unroll
  for (int c2 = 1; c2 < NC_; c2++) mx = fmaxf(mx, src[c2]);
  float s = 0.f;
  #pragma unroll
  for (int c2 = 0; c2 < NC_; c2++) s += expf(src[c2] - mx);
  float lse = mx + logf(s);
  #pragma unroll
  for (int c2 = 0; c2 < NC_; c2++)
    out[(((size_t)b * NC_ + c2) * T_ + t) * 88 + p] = src[c2] - lse;
}

// ---------------- host orchestration ---------------------------------------
extern "C" void kernel_launch(void* const* d_in, const int* in_sizes, int n_in,
                              void* d_out, int out_size, void* d_ws, size_t ws_size,
                              hipStream_t stream) {
  const float* mel     = (const float*)d_in[0];
  const int*   plabel  = (const int*)d_in[1];
  const float* c1w = (const float*)d_in[2];  const float* c1b = (const float*)d_in[3];
  const float* g1  = (const float*)d_in[4];  const float* bb1 = (const float*)d_in[5];
  const float* m1  = (const float*)d_in[6];  const float* v1  = (const float*)d_in[7];
  const float* c2w = (const float*)d_in[8];  const float* c2b = (const float*)d_in[9];
  const float* g2  = (const float*)d_in[10]; const float* bb2 = (const float*)d_in[11];
  const float* m2  = (const float*)d_in[12]; const float* v2  = (const float*)d_in[13];
  const float* c3w = (const float*)d_in[14]; const float* c3b = (const float*)d_in[15];
  const float* g3  = (const float*)d_in[16]; const float* bb3 = (const float*)d_in[17];
  const float* m3  = (const float*)d_in[18]; const float* v3  = (const float*)d_in[19];
  const float* fcw = (const float*)d_in[20]; const float* fcb = (const float*)d_in[21];
  const float* emb = (const float*)d_in[22];
  const float* wih0 = (const float*)d_in[23]; const float* whh0 = (const float*)d_in[24];
  const float* bih0 = (const float*)d_in[25]; const float* bhh0 = (const float*)d_in[26];
  const float* wih1 = (const float*)d_in[27]; const float* whh1 = (const float*)d_in[28];
  const float* bih1 = (const float*)d_in[29]; const float* bhh1 = (const float*)d_in[30];
  const float* outw = (const float*)d_in[31]; const float* outb = (const float*)d_in[32];

  char* ws = (char*)d_ws;
  size_t off = 0;
  auto alloc = [&](size_t bytes) -> char* {
    char* p = ws + off;
    off = (off + bytes + 255) & ~(size_t)255;
    return p;
  };

  _Float16* x0     = (_Float16*)alloc((size_t)BT_ * MEL_ * 2);
  _Float16* buf1   = (_Float16*)alloc((size_t)BT_ * MEL_ * CU_ * 2);  // out1, later out3
  _Float16* buf2   = (_Float16*)alloc((size_t)BT_ * MEL_ * CU_ * 2);  // out2, later A_fc
  _Float16* pool2b = (_Float16*)alloc((size_t)BT_ * MW2_ * CU_ * 2);
  _Float16* ac     = (_Float16*)alloc((size_t)BT_ * FCN_ * 2);
  _Float16* comb   = (_Float16*)alloc((size_t)BT_ * KIN0_ * 2);
  float*    gx     = (float*)   alloc((size_t)BT_ * G4_ * 4);
  _Float16* h1s    = (_Float16*)alloc((size_t)BT_ * H_ * 2);
  _Float16* h2s    = (_Float16*)alloc((size_t)BT_ * H_ * 2);
  float*    logits = (float*)   alloc((size_t)BT_ * NOUT_ * 4);
  _Float16* fcWh   = (_Float16*)alloc((size_t)FCN_ * KFC_ * 2);
  _Float16* wih0h  = (_Float16*)alloc((size_t)G4_ * KIN0_ * 2);
  _Float16* whh0h  = (_Float16*)alloc((size_t)G4_ * H_ * 2);
  _Float16* wih1h  = (_Float16*)alloc((size_t)G4_ * H_ * 2);
  _Float16* whh1h  = (_Float16*)alloc((size_t)G4_ * H_ * 2);
  _Float16* outwh  = (_Float16*)alloc((size_t)NOUT_ * H_ * 2);
  _Float16* w1e    = (_Float16*)alloc(CU_ * 9 * 2);
  float*    b1e    = (float*)   alloc(CU_ * 4);
  _Float16* w2e    = (_Float16*)alloc(CU_ * KCONV_ * 2);
  float*    b2e    = (float*)   alloc(CU_ * 4);
  _Float16* w3e    = (_Float16*)alloc(C3_ * KCONV_ * 2);
  float*    b3e    = (float*)   alloc(C3_ * 4);
  float*    bs0    = (float*)   alloc(G4_ * 4);
  float*    bs1    = (float*)   alloc(G4_ * 4);
  _Float16* hpadA  = (_Float16*)alloc(16 * H_ * 2);
  _Float16* hpadB  = (_Float16*)alloc(16 * H_ * 2);
  float*    cst    = (float*)   alloc(B_ * H_ * 4);
  _Float16* out3   = buf1;   // alias: conv1 output dead after conv2
  _Float16* A_fc   = buf2;   // alias: conv2 output dead after pool2

  auto blocks = [](long long n) { return (unsigned)((n + 255) / 256); };
  auto gemmG = [&](const _Float16* A, const _Float16* W, const float* bias,
                   float* oF, _Float16* oH, int Mr, int Nn, int Kk, int relu) {
    dim3 grid((Nn + BN - 1) / BN, (Mr + BM - 1) / BM);
    wmma_gemm_kernel<false><<<grid, 256, 0, stream>>>(A, W, bias, oF, oH, Mr, Nn, Kk,
                                                      relu, nullptr, 0, 0, 0);
  };
  auto gemmConv = [&](const _Float16* W, const float* bias, _Float16* oH,
                      int Mr, int Nn, const _Float16* cin, int cT, int cMw, int Ci) {
    dim3 grid((Nn + BN - 1) / BN, (Mr + BM - 1) / BM);
    wmma_gemm_kernel<true><<<grid, 256, 0, stream>>>(nullptr, W, bias, nullptr, oH, Mr, Nn,
                                                     KCONV_, 1, cin, cT, cMw, Ci);
  };
  auto gemmAsync = [&](const _Float16* A, const _Float16* W, const float* bias,
                       float* oF, _Float16* oH, int Mr, int Nn, int Kk, int relu) {
    dim3 grid(Nn / BN, Mr / BM);   // aligned: Mr%128==0, Nn%64==0, Kk%32==0
    wmma_gemm_async_kernel<<<grid, 256, 0, stream>>>(A, W, bias, oF, oH, Mr, Nn, Kk, relu);
  };

  // ---- weight prep (fold BN, fp32->fp16, K-pad)
  logmel_kernel<<<blocks((long long)BT_ * MEL_), 256, 0, stream>>>(mel, x0);
  conv_prep_kernel<<<blocks(CU_ * 9), 256, 0, stream>>>(c1w, c1b, g1, bb1, m1, v1, w1e, b1e, CU_, 1, 9);
  conv_prep_kernel<<<blocks(CU_ * KCONV_), 256, 0, stream>>>(c2w, c2b, g2, bb2, m2, v2, w2e, b2e, CU_, CU_, KCONV_);
  conv_prep_kernel<<<blocks(C3_ * KCONV_), 256, 0, stream>>>(c3w, c3b, g3, bb3, m3, v3, w3e, b3e, C3_, CU_, KCONV_);
  cvtpad_kernel<<<blocks((long long)FCN_ * KFC_), 256, 0, stream>>>(fcw, fcWh, FCN_, KFC_, KFC_);
  cvtpad_kernel<<<blocks((long long)G4_ * KIN0_), 256, 0, stream>>>(wih0, wih0h, G4_, 2480, KIN0_);
  cvtpad_kernel<<<blocks((long long)G4_ * H_), 256, 0, stream>>>(whh0, whh0h, G4_, H_, H_);
  cvtpad_kernel<<<blocks((long long)G4_ * H_), 256, 0, stream>>>(wih1, wih1h, G4_, H_, H_);
  cvtpad_kernel<<<blocks((long long)G4_ * H_), 256, 0, stream>>>(whh1, whh1h, G4_, H_, H_);
  cvtpad_kernel<<<blocks((long long)NOUT_ * H_), 256, 0, stream>>>(outw, outwh, NOUT_, H_, H_);
  biassum_kernel<<<blocks(G4_), 256, 0, stream>>>(bih0, bhh0, bs0, G4_);
  biassum_kernel<<<blocks(G4_), 256, 0, stream>>>(bih1, bhh1, bs1, G4_);

  // ---- conv stack
  conv1_kernel<<<blocks((long long)BT_ * MEL_ * CU_), 256, 0, stream>>>(x0, w1e, b1e, buf1);
  gemmConv(w2e, b2e, buf2, BT_ * MEL_, CU_, buf1, T_, MEL_, CU_);
  pool2_kernel<<<blocks((long long)BT_ * MW2_ * CU_), 256, 0, stream>>>(buf2, pool2b);
  gemmConv(w3e, b3e, out3, BT_ * MW2_, C3_, pool2b, T_, MW2_, CU_);
  pool3_kernel<<<blocks((long long)BT_ * KFC_), 256, 0, stream>>>(out3, A_fc);

  // ---- FC + context stacking + embedding concat
  gemmAsync(A_fc, fcWh, fcb, nullptr, ac, BT_, FCN_, KFC_, 0);
  comb_kernel<<<blocks((long long)BT_ * KIN0_), 256, 0, stream>>>(plabel, emb, ac, comb);

  // ---- LSTM layer 0
  gemmAsync(comb, wih0h, bs0, gx, nullptr, BT_, G4_, KIN0_, 0);
  zero_kernel<<<blocks(16 * H_ / 2), 256, 0, stream>>>((unsigned int*)hpadA, 16 * H_ / 2);
  zero_kernel<<<blocks(16 * H_ / 2), 256, 0, stream>>>((unsigned int*)hpadB, 16 * H_ / 2);
  zero_kernel<<<blocks(B_ * H_), 256, 0, stream>>>((unsigned int*)cst, B_ * H_);
  for (int t = 0; t < T_; ++t) {
    const _Float16* hin = (t & 1) ? hpadB : hpadA;
    _Float16* hout = (t & 1) ? hpadA : hpadB;
    lstm_step_kernel<<<H_ / 128, 256, 0, stream>>>(hin, whh0h, gx, cst, hout, h1s, t);
  }

  // ---- LSTM layer 1
  gemmAsync(h1s, wih1h, bs1, gx, nullptr, BT_, G4_, H_, 0);
  zero_kernel<<<blocks(16 * H_ / 2), 256, 0, stream>>>((unsigned int*)hpadA, 16 * H_ / 2);
  zero_kernel<<<blocks(16 * H_ / 2), 256, 0, stream>>>((unsigned int*)hpadB, 16 * H_ / 2);
  zero_kernel<<<blocks(B_ * H_), 256, 0, stream>>>((unsigned int*)cst, B_ * H_);
  for (int t = 0; t < T_; ++t) {
    const _Float16* hin = (t & 1) ? hpadB : hpadA;
    _Float16* hout = (t & 1) ? hpadA : hpadB;
    lstm_step_kernel<<<H_ / 128, 256, 0, stream>>>(hin, whh1h, gx, cst, hout, h2s, t);
  }

  // ---- output projection + log-softmax over NC
  gemmG(h2s, outwh, outb, logits, nullptr, BT_, NOUT_, H_, 0);
  logsoftmax_kernel<<<blocks((long long)BT_ * 88), 256, 0, stream>>>(logits, (float*)d_out);
}
